// AttentionNet_55233279426945
// MI455X (gfx1250) — matrix operand order repaired and checked by reading, not compiled, not run on hardware
//
#include <hip/hip_runtime.h>
#include <hip/hip_bf16.h>

#define A_    8
#define NC_   64
#define OBS_  64
#define POI_  32
#define HID_  256
#define H_    2
#define D_    128
#define B_    4096
#define FEAT_ 2560
#define RSQRT_D 0.08838834764831845f  // 1/sqrt(128)

typedef __attribute__((ext_vector_type(16))) _Float16 v16h;
typedef __attribute__((ext_vector_type(8)))  float    v8f;
typedef __attribute__((ext_vector_type(4)))  unsigned int v4u;

union Frag16 { v16h h; v4u u[2]; };

// ---- WMMA helpers -----------------------------------------------------------
// A-matrix 16x32 f16 layout (ISA 7.12.2): lane = g*16+m; halves j=0..7 -> K=kb+g*8+j,
// j=8..15 -> K=kb+16+g*8+(j-8). Two contiguous 16B chunks from a row-major LDS tile.
__device__ inline v16h load_a_frag(const _Float16* tile, int ldk, int kb) {
  int lane = threadIdx.x & 31;
  int g = lane >> 4, m = lane & 15;
  const _Float16* p = tile + m * ldk + kb + g * 8;
  Frag16 f;
  f.u[0] = *(const v4u*)(p);
  f.u[1] = *(const v4u*)(p + 16);
  return f.h;
}
// B-matrix 32x16 f16: weights prepacked as [N][K] row-major f16; lane n = n0+(lane&15),
// halves j=0..7 -> K = kb + g*8 + j, j=8..15 -> K = kb + 16 + g*8 + (j-8) (two global_load_b128).
__device__ inline v16h load_b_frag(const _Float16* wB, int ldk, int n0, int kb) {
  int lane = threadIdx.x & 31;
  int g = lane >> 4, n = n0 + (lane & 15);
  const _Float16* p = wB + (size_t)n * ldk + kb + g * 8;
  Frag16 f;
  f.u[0] = *(const v4u*)(p);
  f.u[1] = *(const v4u*)(p + 16);
  return f.h;
}
__device__ inline v8f wmma_f16(v16h a, v16h b, v8f c) {
  return __builtin_amdgcn_wmma_f32_16x16x32_f16(false, a, false, b, (short)0, c, false, false);
}
// C/D layout: element i of v8f -> M = i + 8*g, N = lane&15
__device__ inline void store_c_f16(_Float16* tile, int ldn, int n0, v8f c) {
  int lane = threadIdx.x & 31;
  int g = lane >> 4, n = n0 + (lane & 15);
#pragma unroll
  for (int i = 0; i < 8; ++i) tile[(i + 8 * g) * ldn + n] = (_Float16)c[i];
}
// lrelu(x) = max(x, 0.01x): one v_mul + one v_max_num_f32 (vs cmp+cndmask)
__device__ inline float lrelu(float x) { return fmaxf(x, 0.01f * x); }

// ---- init: accumulators, argmax slots, cargo mask ---------------------------
__global__ void k_init(const float* __restrict__ obs, float* wsum_nb, float* wsum_p,
                       unsigned long long* slots, float* if_c) {
  int t = threadIdx.x;
  if (t < 16) { wsum_nb[t] = 0.f; wsum_p[t] = 0.f; slots[t] = 0ull; }
  if (t < 64) {
    int idx = (t == 0) ? (NC_ * POI_ - 1) : (POI_ * t - 1);  // (32c-1) mod 2048
    if_c[t] = obs[(size_t)A_ * OBS_ + idx];                  // observations[0,0,512+idx]
  }
}

// ---- weight conversion fp32 -> f16, prepacked [N][K] ------------------------
__global__ void k_prep(const float* __restrict__ W_enc, const float* __restrict__ Wsel_nb,
                       const float* __restrict__ Wk_nb, const float* __restrict__ Wv_nb,
                       const float* __restrict__ Wsel_poi, const float* __restrict__ Wk_poi,
                       _Float16* wenc, _Float16* wsel, _Float16* wkT,
                       _Float16* wv, _Float16* wselp, _Float16* wkpT) {
  int i = blockIdx.x * 256 + threadIdx.x;            // 65536 threads
  if (i < 16384) wenc[i] = (_Float16)W_enc[i];       // [hid][o] already [N][K]
  if (i < 65536) wsel[i] = (_Float16)Wsel_nb[i];     // [h][d][e] -> N=d, K=e
  if (i < 16384) {                                   // wkT[h][o][d] = Wk_nb[h][d][o]
    int h = i >> 13, r = i & 8191, o = r >> 7, d = r & 127;
    wkT[i] = (_Float16)Wk_nb[h * 8192 + d * 64 + o];
  }
  if (i < 16384) wv[i] = (_Float16)Wv_nb[i];         // [h][d][o] -> N=d, K=o
  if (i < 65536) wselp[i] = (_Float16)Wsel_poi[i];   // [h][d'][e]
  if (i < 8192) {                                    // wkpT[h][p][d] = Wk_poi[h][d][p]
    int h = i >> 12, r = i & 4095, p = r >> 7, d = r & 127;
    wkpT[i] = (_Float16)Wk_poi[h * 4096 + d * 32 + p];
  }
}

// ---- K1: enc -> S -> T (all WMMA), per-wave 16-row b-tile -------------------
__global__ __launch_bounds__(128) void k1_selT(const float* __restrict__ obs,
    const float* __restrict__ b_enc, const _Float16* __restrict__ wenc,
    const _Float16* __restrict__ wsel, const _Float16* __restrict__ wkT,
    float* __restrict__ T) {
  __shared__ __align__(16) _Float16 bufE[4][16 * 256];
  __shared__ __align__(16) _Float16 bufS[4][16 * 128];
  int w = threadIdx.x >> 5, lane = threadIdx.x & 31;
  int a = blockIdx.x >> 6;
  int tb = ((blockIdx.x & 63) << 6) + w * 16;
  int g = lane >> 4, nn = lane & 15;
  _Float16* bE = bufE[w];
  _Float16* bS = bufS[w];

  // stage self_in tile (16x64) f32->f16
  for (int idx = lane; idx < 16 * 64; idx += 32) {
    int m = idx >> 6, o = idx & 63;
    bS[idx] = (_Float16)obs[((size_t)a * B_ + tb + m) * FEAT_ + (A_ - 1) * OBS_ + o];
  }
  __syncthreads();
  // enc = lrelu(self_in @ W_enc^T + b): 16x64 * 64x256
  for (int nt = 0; nt < 16; ++nt) {
    v8f acc = {};
    for (int ks = 0; ks < 2; ++ks)
      acc = wmma_f16(load_a_frag(bS, 64, ks * 32), load_b_frag(wenc, 64, nt * 16, ks * 32), acc);
    float bb = b_enc[nt * 16 + nn];
#pragma unroll
    for (int i = 0; i < 8; ++i)
      bE[(i + 8 * g) * 256 + nt * 16 + nn] = (_Float16)lrelu(acc[i] + bb);
  }
  __syncthreads();
  for (int h = 0; h < 2; ++h) {
    // S = enc @ Wsel_nb[h]^T : 16x256 * 256x128
    for (int nt = 0; nt < 8; ++nt) {
      v8f acc = {};
      for (int k = 0; k < 8; ++k)
        acc = wmma_f16(load_a_frag(bE, 256, k * 32),
                       load_b_frag(wsel + h * 32768, 256, nt * 16, k * 32), acc);
      store_c_f16(bS, 128, nt * 16, acc);
    }
    __syncthreads();
    // T = S @ Wk_nb[h] : 16x128 * 128x64
    for (int nt = 0; nt < 4; ++nt) {
      v8f acc = {};
      for (int k = 0; k < 4; ++k)
        acc = wmma_f16(load_a_frag(bS, 128, k * 32),
                       load_b_frag(wkT + h * 8192, 128, nt * 16, k * 32), acc);
#pragma unroll
      for (int i = 0; i < 8; ++i)
        T[(((size_t)h * 8 + a) * B_ + tb + i + 8 * g) * 64 + nt * 16 + nn] = acc[i];
    }
    __syncthreads();
  }
}

// ---- K2: logits = nbd . T / sqrt(D), plus per-(h,a) sum for the mean --------
__global__ __launch_bounds__(256) void k2_logit(const float* __restrict__ obs,
    const float* __restrict__ T, float* __restrict__ lgit, float* __restrict__ wsum_nb) {
  __shared__ float red[256];
  int a = blockIdx.x >> 4;
  int b = ((blockIdx.x & 15) << 8) + threadIdx.x;
  const float* nbr = obs + ((size_t)a * B_ + b) * FEAT_;
  for (int h = 0; h < 2; ++h) {
    const float* tr = T + (((size_t)h * 8 + a) * B_ + b) * 64;
    float tv[64];
#pragma unroll
    for (int o = 0; o < 64; ++o) tv[o] = tr[o];
    float asum = 0.f;
    for (int n = 0; n < 7; ++n) {
      float acc = 0.f;
#pragma unroll
      for (int o = 0; o < 64; ++o) acc += nbr[n * 64 + o] * tv[o];
      acc *= RSQRT_D;
      lgit[(((size_t)h * 8 + a) * B_ + b) * 8 + n] = acc;
      asum += acc;
    }
    red[threadIdx.x] = asum;
    __syncthreads();
    for (int st = 128; st > 0; st >>= 1) {
      if (threadIdx.x < st) red[threadIdx.x] += red[threadIdx.x + st];
      __syncthreads();
    }
    if (threadIdx.x == 0) atomicAdd(&wsum_nb[h * 8 + a], red[0]);
    __syncthreads();
  }
}

// ---- K4: softmax(nb) -> nb_vals(WMMA) -> Sp(WMMA) -> Tp(WMMA) ---------------
__global__ __launch_bounds__(128) void k4_valsSpTp(const float* __restrict__ obs,
    const float* __restrict__ bv_nb, const float* __restrict__ lgit,
    const float* __restrict__ wsum_nb, const _Float16* __restrict__ wv,
    const _Float16* __restrict__ wselp, const _Float16* __restrict__ wkpT,
    float* __restrict__ Tp) {
  __shared__ __align__(16) _Float16 bufE[4][16 * 256];
  __shared__ __align__(16) _Float16 bufS[4][16 * 128];
  __shared__ float wl[4][2][16][8];
  int w = threadIdx.x >> 5, lane = threadIdx.x & 31;
  int a = blockIdx.x >> 6;
  int tb = ((blockIdx.x & 63) << 6) + w * 16;
  int g = lane >> 4, nn = lane & 15;
  _Float16* bE = bufE[w];
  _Float16* bS = bufS[w];

  // per-row softmax over n=7 with global mean normalization
  if (lane < 16) {
    for (int h = 0; h < 2; ++h) {
      float mean = wsum_nb[h * 8 + a] * (1.0f / (B_ * 7)) + 1e-9f;
      const float* lr = lgit + (((size_t)h * 8 + a) * B_ + tb + lane) * 8;
      float v[7], mx = -1e30f;
      for (int n = 0; n < 7; ++n) { v[n] = lr[n] / mean; mx = fmaxf(mx, v[n]); }
      float sden = 0.f;
      for (int n = 0; n < 7; ++n) { v[n] = __expf(v[n] - mx); sden += v[n]; }
      for (int n = 0; n < 7; ++n) wl[w][h][lane][n] = v[n] / sden;
    }
  }
  __syncthreads();

  for (int h = 0; h < 2; ++h) {
    float nbv[8][8];
#pragma unroll
    for (int nt = 0; nt < 8; ++nt)
#pragma unroll
      for (int i = 0; i < 8; ++i) nbv[nt][i] = 0.f;
    for (int n = 0; n < 7; ++n) {
      __syncthreads();
      for (int idx = lane; idx < 16 * 64; idx += 32) {
        int m = idx >> 6, o = idx & 63;
        bS[idx] = (_Float16)obs[((size_t)a * B_ + tb + m) * FEAT_ + n * OBS_ + o];
      }
      __syncthreads();
      for (int nt = 0; nt < 8; ++nt) {
        v8f acc = {};
        for (int ks = 0; ks < 2; ++ks)
          acc = wmma_f16(load_a_frag(bS, 64, ks * 32),
                         load_b_frag(wv + h * 8192, 64, nt * 16, ks * 32), acc);
        float bb = bv_nb[h * 128 + nt * 16 + nn];
#pragma unroll
        for (int i = 0; i < 8; ++i)
          nbv[nt][i] += lrelu(acc[i] + bb) * wl[w][h][i + 8 * g][n];
      }
    }
#pragma unroll
    for (int nt = 0; nt < 8; ++nt)
#pragma unroll
      for (int i = 0; i < 8; ++i)
        bE[(i + 8 * g) * 256 + h * 128 + nt * 16 + nn] = (_Float16)nbv[nt][i];
  }
  __syncthreads();

  for (int hp = 0; hp < 2; ++hp) {
    // Sp = nb_all @ Wsel_poi[hp]^T : 16x256 * 256x128
    for (int nt = 0; nt < 8; ++nt) {
      v8f acc = {};
      for (int k = 0; k < 8; ++k)
        acc = wmma_f16(load_a_frag(bE, 256, k * 32),
                       load_b_frag(wselp + hp * 32768, 256, nt * 16, k * 32), acc);
      store_c_f16(bS, 128, nt * 16, acc);
    }
    __syncthreads();
    // Tp = Sp @ Wk_poi[hp] : 16x128 * 128x32
    for (int nt = 0; nt < 2; ++nt) {
      v8f acc = {};
      for (int k = 0; k < 4; ++k)
        acc = wmma_f16(load_a_frag(bS, 128, k * 32),
                       load_b_frag(wkpT + hp * 4096, 128, nt * 16, k * 32), acc);
#pragma unroll
      for (int i = 0; i < 8; ++i)
        Tp[(((size_t)hp * 8 + a) * B_ + tb + i + 8 * g) * 32 + nt * 16 + nn] = acc[i];
    }
    __syncthreads();
  }
}

// ---- K5: lp = poi . Tp / sqrt(D), block per b, poi row via async-to-LDS -----
__global__ __launch_bounds__(256) void k5_lp(const float* __restrict__ obs,
    const float* __restrict__ Tp, float* __restrict__ lp, float* __restrict__ wsum_p) {
  __shared__ __align__(16) float pois[2048];
  __shared__ float tps[512];
  __shared__ float sums[16];
  int b = blockIdx.x, t = threadIdx.x;
  // Async byte-copy of the 8KB poi row straight into LDS (no VGPR round trip):
  // GLOBAL_LOAD_ASYNC_TO_LDS_B128, GVS mode, tracked with ASYNCcnt.
  {
    const char* gbase = (const char*)(obs + (size_t)b * FEAT_ + A_ * OBS_);
    unsigned ldsbase = (unsigned)(unsigned long long)(void*)pois;
#pragma unroll
    for (int j = 0; j < 2; ++j) {
      unsigned off = (unsigned)(t + 256 * j) * 16u;  // 512 chunks x 16B = 8KB
      unsigned ldsa = ldsbase + off;
      asm volatile("global_load_async_to_lds_b128 %0, %1, %2"
                   :: "v"(ldsa), "v"(off), "s"(gbase) : "memory");
    }
  }
  for (int i = t; i < 512; i += 256) {
    int ha = i >> 5, p = i & 31;
    tps[i] = Tp[((size_t)ha * B_ + b) * 32 + p];
  }
  if (t < 16) sums[t] = 0.f;
  asm volatile("s_wait_asynccnt 0x0" ::: "memory");  // drain this wave's async copies
  __syncthreads();
  const float s = RSQRT_D;
#pragma unroll
  for (int j = 0; j < 4; ++j) {
    int idx = t + 256 * j;
    int ha = idx >> 6, c = idx & 63;
    float acc = 0.f;
#pragma unroll
    for (int p = 0; p < 32; ++p) acc += pois[c * 32 + p] * tps[ha * 32 + p];
    acc *= s;
    lp[((size_t)ha * B_ + b) * 64 + c] = acc;
    atomicAdd(&sums[ha], acc);   // ds_add_f32
  }
  __syncthreads();
  if (t < 16) atomicAdd(&wsum_p[t], sums[t]);
}

// ---- K6: wp = softmax(lp / (mean+1e-9)) over c=64 ---------------------------
__global__ __launch_bounds__(256) void k6_wp(const float* __restrict__ lp,
    const float* __restrict__ wsum_p, float* __restrict__ wp) {
  unsigned int i = blockIdx.x * 256 + threadIdx.x;  // 16*B
  int ha = i >> 12, b = i & 4095;
  float mean = wsum_p[ha] * (1.0f / (B_ * 64)) + 1e-9f;
  const float* lr = lp + ((size_t)ha * B_ + b) * 64;
  float v[64], mx = -1e30f;
  for (int c = 0; c < 64; ++c) { v[c] = lr[c] / mean; mx = fmaxf(mx, v[c]); }
  float sden = 0.f;
  for (int c = 0; c < 64; ++c) { v[c] = __expf(v[c] - mx); sden += v[c]; }
  float inv = 1.f / sden;
  float* wr = wp + ((size_t)ha * B_ + b) * 64;
  for (int c = 0; c < 64; ++c) wr[c] = v[c] * inv;
}

// ---- scan step: masked global argmax via packed u64 atomic max --------------
__global__ __launch_bounds__(256) void k_argmax(const float* __restrict__ wp,
    const float* __restrict__ if_c, unsigned long long* slots, int step) {
  __shared__ unsigned long long red[256];
  const float* row = wp + (size_t)step * B_ * 64;
  unsigned long long best = 0ull;
#pragma unroll
  for (int j = 0; j < 4; ++j) {
    unsigned int flat = blockIdx.x * 256 + threadIdx.x + 65536u * j;
    int c = flat & 63;
    float v = (if_c[c] == 1.0f) ? 0.f : row[flat];
    // value bits high, ~index low: max() => max value, first index on ties (JAX argmax)
    unsigned long long pk = ((unsigned long long)__float_as_uint(v) << 32) |
                            (unsigned long long)(0xFFFFFFFFu - flat);
    best = best > pk ? best : pk;
  }
  red[threadIdx.x] = best;
  __syncthreads();
  for (int st = 128; st > 0; st >>= 1) {
    if (threadIdx.x < st && red[threadIdx.x + st] > red[threadIdx.x])
      red[threadIdx.x] = red[threadIdx.x + st];
    __syncthreads();
  }
  if (threadIdx.x == 0) atomicMax(&slots[step], red[0]);
}

__global__ __launch_bounds__(256) void k_update(const unsigned long long* slots,
    float* if_c, float* __restrict__ out, int step) {
  __shared__ unsigned int sci;
  if (threadIdx.x == 0) {
    unsigned long long v = slots[step];
    unsigned int ci = 0xFFFFFFFFu - (unsigned int)(v & 0xFFFFFFFFull);
    sci = ci;
    if (ci < (unsigned)NC_) if_c[ci] = 1.0f;  // JAX drops OOB scatter updates
  }
  __syncthreads();
  int a = step & 7;
  float cf = (float)sci;
  for (int j = threadIdx.x; j < B_; j += 256) out[(size_t)a * B_ + j] = cf;
}

// ---- launcher ---------------------------------------------------------------
extern "C" void kernel_launch(void* const* d_in, const int* in_sizes, int n_in,
                              void* d_out, int out_size, void* d_ws, size_t ws_size,
                              hipStream_t stream) {
  (void)in_sizes; (void)n_in; (void)out_size; (void)ws_size;
  const float* obs      = (const float*)d_in[0];
  const float* W_enc    = (const float*)d_in[1];
  const float* b_enc    = (const float*)d_in[2];
  const float* Wk_nb    = (const float*)d_in[3];
  const float* Wsel_nb  = (const float*)d_in[4];
  const float* Wv_nb    = (const float*)d_in[5];
  const float* bv_nb    = (const float*)d_in[6];
  const float* Wk_poi   = (const float*)d_in[7];
  const float* Wsel_poi = (const float*)d_in[8];
  // d_in[9]=Wv_poi, d_in[10]=bv_poi are unused by the reference math
  float* out = (float*)d_out;

  float* wsum_nb = (float*)d_ws;                                   // 16
  float* wsum_p  = wsum_nb + 16;                                   // 16
  unsigned long long* slots = (unsigned long long*)(wsum_p + 16);  // 16
  float* if_c = (float*)(slots + 16);                              // 64
  _Float16* wenc  = (_Float16*)(if_c + 64);                        // 16384
  _Float16* wsel  = wenc + 16384;                                  // 65536
  _Float16* wkT   = wsel + 65536;                                  // 16384
  _Float16* wv    = wkT + 16384;                                   // 16384
  _Float16* wselp = wv + 16384;                                    // 65536
  _Float16* wkpT  = wselp + 65536;                                 // 8192
  float* T    = (float*)(wkpT + 8192);                             // 16*4096*64
  float* lgit = T + (size_t)16 * 4096 * 64;                        // 16*4096*8
  float* Tp   = lgit + (size_t)16 * 4096 * 8;                      // 16*4096*32
  float* lp   = Tp + (size_t)16 * 4096 * 32;                       // 16*4096*64
  float* wp   = lp + (size_t)16 * 4096 * 64;                       // 16*4096*64

  k_init<<<1, 256, 0, stream>>>(obs, wsum_nb, wsum_p, slots, if_c);
  k_prep<<<256, 256, 0, stream>>>(W_enc, Wsel_nb, Wk_nb, Wv_nb, Wsel_poi, Wk_poi,
                                  wenc, wsel, wkT, wv, wselp, wkpT);
  k1_selT<<<512, 128, 0, stream>>>(obs, b_enc, wenc, wsel, wkT, T);
  k2_logit<<<128, 256, 0, stream>>>(obs, T, lgit, wsum_nb);
  k4_valsSpTp<<<512, 128, 0, stream>>>(obs, bv_nb, lgit, wsum_nb, wv, wselp, wkpT, Tp);
  k5_lp<<<4096, 256, 0, stream>>>(obs, Tp, lp, wsum_p);
  k6_wp<<<256, 256, 0, stream>>>(lp, wsum_p, wp);
  for (int step = 0; step < 16; ++step) {
    k_argmax<<<256, 256, 0, stream>>>(wp, if_c, slots, step);
    k_update<<<1, 256, 0, stream>>>(slots, if_c, out, step);
  }
}
